// PointNet2Encoder_60610578481762
// MI455X (gfx1250) — matrix operand order repaired and checked by reading, not compiled, not run on hardware
//
#include <hip/hip_runtime.h>
#include <hip/hip_bf16.h>

typedef _Float16 v16h __attribute__((ext_vector_type(16)));
typedef _Float16 v8h  __attribute__((ext_vector_type(8)));
typedef float    v8f  __attribute__((ext_vector_type(8)));

// ------------------------------------------------------------------
// Furthest point sampling: one 1024-thread workgroup per batch.
// Serial over npoint iterations; LDS tree argmax with low-index tiebreak
// (matches jnp.argmax semantics).
// ------------------------------------------------------------------
__global__ __launch_bounds__(1024)
void pn2_fps(const float* __restrict__ support, int N, int npoint,
             float* __restrict__ dist, int* __restrict__ idx_out)
{
  const int b = blockIdx.x;
  const int t = threadIdx.x;
  const float* xyz = support + (size_t)b * N * 3;
  float* d = dist + (size_t)b * N;
  int* idx = idx_out + (size_t)b * npoint;

  __shared__ float s_val[1024];
  __shared__ int   s_idx[1024];
  __shared__ float s_last[3];

  for (int i = t; i < N; i += 1024) d[i] = 1e10f;
  if (t == 0) { idx[0] = 0; s_last[0] = xyz[0]; s_last[1] = xyz[1]; s_last[2] = xyz[2]; }
  __syncthreads();

  for (int it = 1; it < npoint; ++it) {
    const float lx = s_last[0], ly = s_last[1], lz = s_last[2];
    float best = -1.0f; int bi = 0;
    for (int i = t; i < N; i += 1024) {
      float dx = xyz[3*i] - lx, dy = xyz[3*i+1] - ly, dz = xyz[3*i+2] - lz;
      float dd = dx*dx + dy*dy + dz*dz;
      float nd = fminf(d[i], dd);
      d[i] = nd;
      if (nd > best) { best = nd; bi = i; }
    }
    s_val[t] = best; s_idx[t] = bi;
    __syncthreads();
    for (int s = 512; s > 0; s >>= 1) {
      if (t < s) {
        float v2 = s_val[t+s]; int i2 = s_idx[t+s];
        if (v2 > s_val[t] || (v2 == s_val[t] && i2 < s_idx[t])) { s_val[t] = v2; s_idx[t] = i2; }
      }
      __syncthreads();
    }
    if (t == 0) {
      int w = s_idx[0];
      idx[it] = w;
      s_last[0] = xyz[3*w]; s_last[1] = xyz[3*w+1]; s_last[2] = xyz[3*w+2];
    }
    __syncthreads();
  }
}

// feats[b][c][n] = xyz[b][n][c]  (stage-0 feature master)
__global__ void pn2_xyzT(const float* __restrict__ xyz, float* __restrict__ feats,
                         int N, size_t fslab)
{
  int n = blockIdx.x * blockDim.x + threadIdx.x;
  int b = blockIdx.z;
  if (n >= N) return;
  const float* p = xyz + ((size_t)b * N + n) * 3;
  float* f = feats + (size_t)b * fslab;
  f[(size_t)0*N + n] = p[0];
  f[(size_t)1*N + n] = p[1];
  f[(size_t)2*N + n] = p[2];
}

__global__ void pn2_gatherq(const float* __restrict__ support, const int* __restrict__ idx,
                            float* __restrict__ query, int N, int Q)
{
  int q = blockIdx.x * blockDim.x + threadIdx.x;
  int b = blockIdx.z;
  if (q >= Q) return;
  int id = idx[(size_t)b * Q + q];
  const float* s = support + ((size_t)b * N + id) * 3;
  float* o = query + ((size_t)b * Q + q) * 3;
  o[0] = s[0]; o[1] = s[1]; o[2] = s[2];
}

// Ball query: first 32 in-radius neighbors in increasing index order,
// padded with the first hit (PointNet++ CUDA semantics).
__global__ void pn2_ballq(const float* __restrict__ query, const float* __restrict__ support,
                          int* __restrict__ nbr, int N, int Q, float r2)
{
  int q = blockIdx.x * blockDim.x + threadIdx.x;
  int b = blockIdx.z;
  if (q >= Q) return;
  const float* qp = query + ((size_t)b * Q + q) * 3;
  const float qx = qp[0], qy = qp[1], qz = qp[2];
  const float* s = support + (size_t)b * N * 3;
  int* out = nbr + ((size_t)b * Q + q) * 32;
  int cnt = 0, first = 0;
  for (int i = 0; i < N && cnt < 32; ++i) {
    float dx = s[3*i] - qx, dy = s[3*i+1] - qy, dz = s[3*i+2] - qz;
    if (dx*dx + dy*dy + dz*dz < r2) {
      if (cnt == 0) first = i;
      out[cnt++] = i;
    }
  }
  for (; cnt < 32; ++cnt) out[cnt] = first;
}

// Build f16 input matrix G[cin_pad][Q*32]: rows 0..2 = dp, rows 3.. = gathered feats,
// zero-padded K rows for WMMA.
__global__ void pn2_build(const float* __restrict__ query, const float* __restrict__ support,
                          const int* __restrict__ nbr, const float* __restrict__ feats,
                          _Float16* __restrict__ G, int N, int Q, int Cprev, int cin_pad,
                          size_t fslab, size_t gslab)
{
  int j = blockIdx.x * blockDim.x + threadIdx.x;
  int b = blockIdx.z;
  int QS = Q * 32;
  if (j >= QS) return;
  int q = j >> 5;
  int id = nbr[((size_t)b * Q + q) * 32 + (j & 31)];
  const float* sp = support + ((size_t)b * N + id) * 3;
  const float* qp = query   + ((size_t)b * Q + q) * 3;
  _Float16* g = G + (size_t)b * gslab;
  g[(size_t)0*QS + j] = (_Float16)(sp[0] - qp[0]);
  g[(size_t)1*QS + j] = (_Float16)(sp[1] - qp[1]);
  g[(size_t)2*QS + j] = (_Float16)(sp[2] - qp[2]);
  const float* f = feats + (size_t)b * fslab;
  for (int c = 0; c < Cprev; ++c)
    g[(size_t)(3 + c)*QS + j] = (_Float16)f[(size_t)c * N + id];
  for (int c = 3 + Cprev; c < cin_pad; ++c)
    g[(size_t)c*QS + j] = (_Float16)0.f;
}

// Convert f32 weights to K-padded f16, row-major [cout][cin_pad].
__global__ void pn2_convw(const float* __restrict__ Wf, _Float16* __restrict__ Wh,
                          int cout, int cin, int cin_pad)
{
  int i = blockIdx.x * blockDim.x + threadIdx.x;
  int total = cout * cin_pad;
  if (i >= total) return;
  int r = i / cin_pad, c = i % cin_pad;
  Wh[i] = (c < cin) ? (_Float16)Wf[(size_t)r * cin + c] : (_Float16)0.f;
}

// ------------------------------------------------------------------
// WMMA GEMM: Out[cout][QS] = relu(W[cout][K] x Gin[K][QS] + bias), f16 out.
// One wave per 16(M) x 64(N) strip, K loop in steps of 32 (f16 WMMA).
// A frag per ISA 16-bit A 16x32 layout; B frag: lane%16 = N col, halves = K rows
// in A's k order (strided loads); D per ISA 32-bit C/D 16x16 layout.
// ------------------------------------------------------------------
__global__ __launch_bounds__(32)
void pn2_gemm(const _Float16* __restrict__ W, const float* __restrict__ bias,
              const _Float16* __restrict__ Gin, _Float16* __restrict__ Out,
              int K, int QS, size_t slab_in, size_t slab_out)
{
  const int lane = threadIdx.x;
  const int n0 = blockIdx.x * 64;
  const int m0 = blockIdx.y * 16;
  const int b  = blockIdx.z;
  const _Float16* Gb = Gin + (size_t)b * slab_in;
  _Float16* Ob = Out + (size_t)b * slab_out;

  const int hi = lane >> 4;
  const int lm = lane & 15;

  v8f acc0 = {}, acc1 = {}, acc2 = {}, acc3 = {};
  const _Float16* aRow = W + (size_t)(m0 + lm) * K + 8 * hi;
  const _Float16* bRow = Gb + (size_t)(8 * hi) * QS + n0 + lm;

  for (int k0 = 0; k0 < K; k0 += 32) {
    v8h alo = *(const v8h*)(aRow + k0);
    v8h ahi = *(const v8h*)(aRow + k0 + 16);
    v16h a;
#pragma unroll
    for (int i = 0; i < 8; ++i) { a[i] = alo[i]; a[8 + i] = ahi[i]; }
    const _Float16* bp = bRow + (size_t)k0 * QS;
    v16h b0, b1, b2, b3;
#pragma unroll
    for (int i = 0; i < 8; ++i) {
      const _Float16* pl = bp + (size_t)i * QS;
      const _Float16* ph = bp + (size_t)(16 + i) * QS;
      b0[i] = pl[ 0]; b0[8 + i] = ph[ 0];
      b1[i] = pl[16]; b1[8 + i] = ph[16];
      b2[i] = pl[32]; b2[8 + i] = ph[32];
      b3[i] = pl[48]; b3[8 + i] = ph[48];
    }
    acc0 = __builtin_amdgcn_wmma_f32_16x16x32_f16(false, a, false, b0, (short)0, acc0, false, false);
    acc1 = __builtin_amdgcn_wmma_f32_16x16x32_f16(false, a, false, b1, (short)0, acc1, false, false);
    acc2 = __builtin_amdgcn_wmma_f32_16x16x32_f16(false, a, false, b2, (short)0, acc2, false, false);
    acc3 = __builtin_amdgcn_wmma_f32_16x16x32_f16(false, a, false, b3, (short)0, acc3, false, false);
  }

#pragma unroll
  for (int r = 0; r < 8; ++r) {
    const int row = m0 + r + 8 * hi;
    const float bv = bias[row];
    const size_t rb = (size_t)row * QS + (size_t)n0 + lm;
    float v;
    v = acc0[r] + bv; Ob[rb +  0] = (_Float16)(v > 0.f ? v : 0.f);
    v = acc1[r] + bv; Ob[rb + 16] = (_Float16)(v > 0.f ? v : 0.f);
    v = acc2[r] + bv; Ob[rb + 32] = (_Float16)(v > 0.f ? v : 0.f);
    v = acc3[r] + bv; Ob[rb + 48] = (_Float16)(v > 0.f ? v : 0.f);
  }
}

// Max over 32 neighbors: feats_out[b][c][q] = max_s g[b][c][q*32+s]
__global__ void pn2_maxpool(const _Float16* __restrict__ Gin, float* __restrict__ feats_out,
                            int C, int Q, size_t gslab, size_t fslab)
{
  int cq = blockIdx.x * blockDim.x + threadIdx.x;
  int b = blockIdx.z;
  if (cq >= C * Q) return;
  int c = cq / Q, q = cq % Q;
  const _Float16* g = Gin + (size_t)b * gslab + (size_t)c * Q * 32 + (size_t)q * 32;
  float m = (float)g[0];
  for (int s = 1; s < 32; ++s) m = fmaxf(m, (float)g[s]);
  feats_out[(size_t)b * fslab + (size_t)c * Q + q] = m;
}

// ------------------------------------------------------------------
extern "C" void kernel_launch(void* const* d_in, const int* in_sizes, int n_in,
                              void* d_out, int out_size, void* d_ws, size_t ws_size,
                              hipStream_t stream)
{
  (void)in_sizes; (void)n_in; (void)out_size; (void)ws_size;
  const int B = 2;
  const float* xyz = (const float*)d_in[0];

  char* base = (char*)d_ws;
  size_t off = 0;
  auto alloc = [&](size_t bytes) -> char* {
    char* p = base + off;
    off = (off + bytes + 255) & ~(size_t)255;
    return p;
  };

  float*    dist   = (float*)alloc((size_t)B * 16384 * 4);
  int*      idxbuf = (int*)  alloc((size_t)B * 4096 * 4);
  float*    qxyzA  = (float*)alloc((size_t)B * 4096 * 3 * 4);
  float*    qxyzB  = (float*)alloc((size_t)B * 4096 * 3 * 4);
  int*      nbr    = (int*)  alloc((size_t)B * 4096 * 32 * 4);
  const size_t FSLAB = 524288;       // max C*N floats per batch (128*4096)
  float*    featsA = (float*)alloc((size_t)B * FSLAB * 4);
  float*    featsB = (float*)alloc((size_t)B * FSLAB * 4);
  _Float16* Wh     = (_Float16*)alloc((size_t)1419264 * 2);
  const size_t GSLAB  = 5242880;     // max cin_pad*QS halves (160*32768)
  const size_t H1SLAB = 8388608;     // max c1*QS (64*131072)
  const size_t H2SLAB = 8388608;
  const size_t H3SLAB = 16777216;    // max c3*QS (128*131072)
  _Float16* G  = (_Float16*)alloc((size_t)B * GSLAB  * 2);
  _Float16* H1 = (_Float16*)alloc((size_t)B * H1SLAB * 2);
  _Float16* H2 = (_Float16*)alloc((size_t)B * H2SLAB * 2);
  _Float16* H3 = (_Float16*)alloc((size_t)B * H3SLAB * 2);

  static const int couts[12] = {64,64,128, 128,128,256, 256,256,512, 512,512,1024};
  static const int cins [12] = {6,64,64, 131,128,128, 259,256,256, 515,512,512};
  static const int kpads[12] = {32,64,64, 160,128,128, 288,256,256, 544,512,512};
  size_t woff[12]; size_t wacc = 0;
  for (int i = 0; i < 12; ++i) { woff[i] = wacc; wacc += (size_t)couts[i] * kpads[i]; }
  for (int i = 0; i < 12; ++i) {
    int total = couts[i] * kpads[i];
    pn2_convw<<<dim3((total + 255) / 256), dim3(256), 0, stream>>>(
        (const float*)d_in[1 + 2*i], Wh + woff[i], couts[i], cins[i], kpads[i]);
  }

  pn2_xyzT<<<dim3(16384/256, 1, B), dim3(256), 0, stream>>>(xyz, featsA, 16384, FSLAB);

  struct Cfg { int N, Q, Cprev, cin_pad, c1, c2, c3; float r; };
  const Cfg cfg[4] = {
    {16384, 4096,   3,  32,  64,  64,  128, 0.1f},
    { 4096, 1024, 128, 160, 128, 128,  256, 0.2f},
    { 1024,  256, 256, 288, 256, 256,  512, 0.4f},
    {  256,   64, 512, 544, 512, 512, 1024, 0.8f},
  };

  const float* support = xyz;
  float* qcur = qxyzA; float* qalt = qxyzB;
  float* fprev = featsA; float* fnext = featsB;

  for (int k = 0; k < 4; ++k) {
    const Cfg c = cfg[k];
    const int QS = c.Q * 32;

    pn2_fps<<<dim3(B), dim3(1024), 0, stream>>>(support, c.N, c.Q, dist, idxbuf);
    pn2_gatherq<<<dim3((c.Q + 255) / 256, 1, B), dim3(256), 0, stream>>>(
        support, idxbuf, qcur, c.N, c.Q);
    pn2_ballq<<<dim3((c.Q + 255) / 256, 1, B), dim3(256), 0, stream>>>(
        qcur, support, nbr, c.N, c.Q, c.r * c.r);
    pn2_build<<<dim3(QS / 256, 1, B), dim3(256), 0, stream>>>(
        qcur, support, nbr, fprev, G, c.N, c.Q, c.Cprev, c.cin_pad, FSLAB, GSLAB);

    const float* b0 = (const float*)d_in[2 + 2*(3*k + 0)];
    const float* b1 = (const float*)d_in[2 + 2*(3*k + 1)];
    const float* b2 = (const float*)d_in[2 + 2*(3*k + 2)];

    pn2_gemm<<<dim3(QS/64, c.c1/16, B), dim3(32), 0, stream>>>(
        Wh + woff[3*k + 0], b0, G,  H1, c.cin_pad, QS, GSLAB,  H1SLAB);
    pn2_gemm<<<dim3(QS/64, c.c2/16, B), dim3(32), 0, stream>>>(
        Wh + woff[3*k + 1], b1, H1, H2, c.c1,      QS, H1SLAB, H2SLAB);
    pn2_gemm<<<dim3(QS/64, c.c3/16, B), dim3(32), 0, stream>>>(
        Wh + woff[3*k + 2], b2, H2, H3, c.c2,      QS, H2SLAB, H3SLAB);

    float* outF = (k == 3) ? (float*)d_out : fnext;
    size_t fslab = (k == 3) ? (size_t)(1024 * 64) : FSLAB;
    pn2_maxpool<<<dim3((c.c3 * c.Q + 255) / 256, 1, B), dim3(256), 0, stream>>>(
        H3, outF, c.c3, c.Q, H3SLAB, fslab);

    support = qcur; { float* t = qcur; qcur = qalt; qalt = t; }
    { float* t = fprev; fprev = fnext; fnext = t; }
  }
}
